// DecoderBlock_89060441850075
// MI455X (gfx1250) — matrix operand order repaired and checked by reading, not compiled, not run on hardware
//
#include <hip/hip_runtime.h>

// ---------------------------------------------------------------------------
// Types for CDNA5 WMMA (wave32): 16x16x32 bf16 -> f32
// ---------------------------------------------------------------------------
typedef __attribute__((ext_vector_type(16))) __bf16 v16bf;
typedef __attribute__((ext_vector_type(8)))  float  v8f;

union FragB16 { v16bf v; unsigned u[8]; unsigned short us[16]; };
union FragF32 { v8f   v; float f[8]; };

__device__ __forceinline__ unsigned short f2bf(float f) {
    unsigned u = __float_as_uint(f);
    u += 0x7FFFu + ((u >> 16) & 1u);          // round-to-nearest-even
    return (unsigned short)(u >> 16);
}

// ---------------------------------------------------------------------------
// Generic bf16 WMMA GEMM: out = epilogue(A[M,K](bf16) @ W[K,N](f32->bf16))
// Block tile 128x128 (4 waves, each 64x64 -> 16 WMMAs per K-step of 32).
// Staging is two-phase (batch all global loads, then LDS stores) so loads
// overlap under a single wait instead of serializing.
// flags: bit0 = bf16 out, bit1 = exact GELU. bias/residual nullable.
// M, N multiples of 128; K multiple of 32.
// ---------------------------------------------------------------------------
#define BM 128
#define BN 128
#define BK 32

__global__ __launch_bounds__(128) void gemm_bf16_wmma(
    const unsigned short* __restrict__ A, const float* __restrict__ W,
    const float* __restrict__ bias, const float* __restrict__ residual,
    void* __restrict__ out, int M, int N, int K, int flags)
{
    __shared__ __align__(16) unsigned short As[BM][BK];  // [m][k]
    __shared__ __align__(16) unsigned short Bs[BN][BK];  // [n][k] K-contig

    const int t    = threadIdx.x;
    const int w    = t >> 5;
    const int lane = t & 31;
    const int half = lane >> 4;
    const int ln   = lane & 15;
    const int mb   = blockIdx.y * BM;
    const int nb   = blockIdx.x * BN;
    const int wm   = (w & 1) * 64;
    const int wn   = (w >> 1) * 64;

    FragF32 acc[4][4];
    #pragma unroll
    for (int i = 0; i < 4; i++)
        #pragma unroll
        for (int j = 0; j < 4; j++)
            #pragma unroll
            for (int r = 0; r < 8; r++) acc[i][j].f[r] = 0.f;

    for (int k0 = 0; k0 < K; k0 += BK) {
        // ---- phase 1: batch ALL global loads (issued back-to-back) ----
        uint4 av[4];                       // A: 4 x b128 per thread
        #pragma unroll
        for (int j = 0; j < 4; j++) {
            int g   = t + 128 * j;         // 0..511 uint4 units
            int row = g >> 2;              // 4 uint4 per 32-elem row
            int qp  = g & 3;
            av[j] = *((const uint4*)(A + (size_t)(mb + row) * K + k0) + qp);
        }
        float fb[32];                      // B: one 32-deep column, coalesced
        {
            const float* wp = W + (size_t)k0 * N + (nb + t);
            #pragma unroll
            for (int j = 0; j < 32; j++) fb[j] = wp[(size_t)j * N];
        }
        // prefetch next K tile (speculative, dropped if OOB)
        if (k0 + BK < K) {
            __builtin_prefetch(A + (size_t)(mb + (t >> 4)) * K + k0 + BK, 0, 3);
            __builtin_prefetch(W + (size_t)(k0 + BK) * N + (nb + t), 0, 3);
        }
        // ---- phase 2: LDS stores ----
        #pragma unroll
        for (int j = 0; j < 4; j++) {
            int g   = t + 128 * j;
            int row = g >> 2;
            int qp  = g & 3;
            *((uint4*)&As[row][0] + qp) = av[j];
        }
        #pragma unroll
        for (int j = 0; j < 16; j++) {
            unsigned pk = (unsigned)f2bf(fb[2 * j]) |
                          ((unsigned)f2bf(fb[2 * j + 1]) << 16);
            ((unsigned*)&Bs[t][0])[j] = pk;
        }
        __syncthreads();

        // ---- fragments per documented CDNA5 VGPR layouts ----
        FragB16 af[4];
        #pragma unroll
        for (int tr = 0; tr < 4; tr++) {
            int row = wm + tr * 16 + ln;
            #pragma unroll
            for (int v = 0; v < 8; v++) {
                int k = ((v >> 2) << 4) + half * 8 + ((v & 3) << 1);
                af[tr].u[v] = *(const unsigned*)&As[row][k];
            }
        }
        #pragma unroll
        for (int tc = 0; tc < 4; tc++) {
            FragB16 bfr;
            int col = wn + tc * 16 + ln;
            #pragma unroll
            for (int v = 0; v < 8; v++) {
                int k = ((v >> 2) << 4) + half * 8 + ((v & 3) << 1);
                bfr.u[v] = *(const unsigned*)&Bs[col][k];
            }
            #pragma unroll
            for (int tr = 0; tr < 4; tr++)
                acc[tr][tc].v = __builtin_amdgcn_wmma_f32_16x16x32_bf16(
                    false, af[tr].v, false, bfr.v,
                    (short)0, acc[tr][tc].v, false, false);
        }
        __syncthreads();
    }

    const bool obf  = (flags & 1) != 0;
    const bool gelu = (flags & 2) != 0;
    #pragma unroll
    for (int tr = 0; tr < 4; tr++)
        #pragma unroll
        for (int tc = 0; tc < 4; tc++)
            #pragma unroll
            for (int r = 0; r < 8; r++) {
                int row = mb + wm + tr * 16 + r + half * 8;   // C layout M
                int col = nb + wn + tc * 16 + ln;             // C layout N
                float vacc = acc[tr][tc].f[r];
                if (bias)     vacc += bias[col];
                if (gelu)     vacc = 0.5f * vacc * (1.f + erff(vacc * 0.70710678f));
                if (residual) vacc += residual[(size_t)row * N + col];
                if (obf) ((unsigned short*)out)[(size_t)row * N + col] = f2bf(vacc);
                else     ((float*)out)[(size_t)row * N + col]          = vacc;
            }
}

// ---------------------------------------------------------------------------
// Flash attention: 1 wave per 16-row q tile. qkv is bf16 (token-major rows of
// 3072: [q|k|v]). Online softmax, QK^T and PV via WMMA bf16, d=64, S=2048.
// Per key block (32 keys): K and V tiles staged to LDS with batched b128
// loads; fragments built from LDS (same-wave LDS is in-order -> no barrier).
// ---------------------------------------------------------------------------
__global__ __launch_bounds__(128) void attn_wmma(
    const unsigned short* __restrict__ qkv, unsigned short* __restrict__ outb)
{
    __shared__ __align__(16) unsigned short kvs[4][2][32 * 64]; // [wave][K|V]
    __shared__ __align__(16) unsigned short pl[4][16 * 32];     // prob tiles
    const int t    = threadIdx.x;
    const int w    = t >> 5;
    const int lane = t & 31;
    const int half = lane >> 4;
    const int ln   = lane & 15;

    const int tile = blockIdx.x * 4 + w;        // 0..4095
    const int head = tile & 15;
    const int qt   = (tile >> 4) & 127;
    const int b    = tile >> 11;
    const int s0   = b * 2048 + qt * 16;
    const int LD   = 3072;
    const int hoff = head * 64;

    // Q fragments (16 x 64 split into two 16x32 A fragments)
    FragB16 qf[2];
    #pragma unroll
    for (int ks = 0; ks < 2; ks++) {
        int srow = s0 + ln;
        #pragma unroll
        for (int v = 0; v < 8; v++) {
            int k = ks * 32 + ((v >> 2) << 4) + half * 8 + ((v & 3) << 1);
            qf[ks].u[v] = *(const unsigned*)(qkv + (size_t)srow * LD + hoff + k);
        }
    }

    FragF32 o[4];
    #pragma unroll
    for (int dt = 0; dt < 4; dt++)
        #pragma unroll
        for (int r = 0; r < 8; r++) o[dt].f[r] = 0.f;
    float mst[8], lst[8];
    #pragma unroll
    for (int r = 0; r < 8; r++) { mst[r] = -3.0e38f; lst[r] = 0.f; }

    unsigned short* ksm = &kvs[w][0][0];
    unsigned short* vsm = &kvs[w][1][0];
    unsigned short* plw = &pl[w][0];

    for (int kb = 0; kb < 64; kb++) {
        const int sk = b * 2048 + kb * 32;

        // ---- stage K and V tiles: batch 16 b128 loads, then 16 stores ----
        uint4 kq[8], vq[8];
        #pragma unroll
        for (int j = 0; j < 8; j++) {
            int g   = lane + 32 * j;            // 0..255 uint4's
            int ki  = g >> 3;                   // key index 0..31
            int cb  = (g & 7) * 8;              // ushort offset in row
            const unsigned short* base = qkv + (size_t)(sk + ki) * LD + hoff + cb;
            kq[j] = *(const uint4*)(base + 1024);
            vq[j] = *(const uint4*)(base + 2048);
        }
        if (kb < 63) {
            __builtin_prefetch(qkv + (size_t)(sk + 32 + lane) * LD + 1024 + hoff, 0, 3);
            __builtin_prefetch(qkv + (size_t)(sk + 32 + lane) * LD + 2048 + hoff, 0, 3);
        }
        #pragma unroll
        for (int j = 0; j < 8; j++) {
            int g   = lane + 32 * j;
            int ki  = g >> 3;
            int cb  = (g & 7) * 8;
            *(uint4*)&ksm[ki * 64 + cb] = kq[j];
            *(uint4*)&vsm[ki * 64 + cb] = vq[j];
        }

        // ---- scores: S = Q @ K^T (2 N-tiles x 2 K-steps) ----
        FragF32 sc[2];
        #pragma unroll
        for (int nt = 0; nt < 2; nt++) {
            #pragma unroll
            for (int r = 0; r < 8; r++) sc[nt].f[r] = 0.f;
            #pragma unroll
            for (int ks = 0; ks < 2; ks++) {
                FragB16 kf;
                const unsigned short* kp = &ksm[(nt * 16 + ln) * 64 + ks * 32];
                #pragma unroll
                for (int v = 0; v < 8; v++) {
                    int kk = ((v >> 2) << 4) + half * 8 + ((v & 3) << 1);
                    kf.u[v] = *(const unsigned*)(kp + kk);
                }
                sc[nt].v = __builtin_amdgcn_wmma_f32_16x16x32_bf16(
                    false, qf[ks].v, false, kf.v, (short)0, sc[nt].v, false, false);
            }
        }

        // ---- online softmax across the 32 keys of this block ----
        #pragma unroll
        for (int r = 0; r < 8; r++) {
            float s0v = sc[0].f[r] * 0.125f;    // 1/sqrt(64)
            float s1v = sc[1].f[r] * 0.125f;
            float mv  = fmaxf(s0v, s1v);
            mv = fmaxf(mv, __shfl_xor(mv, 1, 32));
            mv = fmaxf(mv, __shfl_xor(mv, 2, 32));
            mv = fmaxf(mv, __shfl_xor(mv, 4, 32));
            mv = fmaxf(mv, __shfl_xor(mv, 8, 32));
            float mnew  = fmaxf(mst[r], mv);
            float alpha = __expf(mst[r] - mnew);
            float p0 = __expf(s0v - mnew);
            float p1 = __expf(s1v - mnew);
            float rs = p0 + p1;
            rs += __shfl_xor(rs, 1, 32);
            rs += __shfl_xor(rs, 2, 32);
            rs += __shfl_xor(rs, 4, 32);
            rs += __shfl_xor(rs, 8, 32);
            lst[r] = lst[r] * alpha + rs;
            mst[r] = mnew;
            #pragma unroll
            for (int dt = 0; dt < 4; dt++) o[dt].f[r] *= alpha;
            int rowm = r + half * 8;            // C-layout row
            plw[rowm * 32 + ln]      = f2bf(p0);
            plw[rowm * 32 + 16 + ln] = f2bf(p1);
        }

        // ---- re-layout probs C->A through LDS ----
        FragB16 pf;
        #pragma unroll
        for (int v = 0; v < 8; v++) {
            int k = half * 16 + (v << 1);
            pf.u[v] = *(const unsigned*)&plw[ln * 32 + k];
        }

        // ---- O += P @ V (four 16x16 d tiles) ----
        #pragma unroll
        for (int dt = 0; dt < 4; dt++) {
            FragB16 vf;
            #pragma unroll
            for (int v = 0; v < 8; v++) {
                int kk  = half * 16 + (v << 1);
                unsigned lo = vsm[kk * 64 + dt * 16 + ln];
                unsigned hi = vsm[(kk + 1) * 64 + dt * 16 + ln];
                vf.u[v] = lo | (hi << 16);
            }
            o[dt].v = __builtin_amdgcn_wmma_f32_16x16x32_bf16(
                false, pf.v, false, vf.v, (short)0, o[dt].v, false, false);
        }
    }

    // ---- normalize and store (B,S,H,d) flattened to (B,S,1024), bf16 ----
    #pragma unroll
    for (int r = 0; r < 8; r++) {
        float inv = 1.f / lst[r];
        int token = s0 + r + half * 8;
        #pragma unroll
        for (int dt = 0; dt < 4; dt++)
            outb[(size_t)token * 1024 + hoff + dt * 16 + ln] =
                f2bf(o[dt].f[r] * inv);
    }
}

// ---------------------------------------------------------------------------
// AdaLN: per-row LayerNorm(no affine) then (1+gamma)*xn + beta -> bf16
// gb rows: [gamma(1024) | beta(1024)]. One block (256 thr) per row of 1024.
// ---------------------------------------------------------------------------
__global__ __launch_bounds__(256) void adaln_kernel(
    const float* __restrict__ x, const float* __restrict__ gb,
    unsigned short* __restrict__ h)
{
    __shared__ float red[16];
    const int row = blockIdx.x;
    const int t   = threadIdx.x;
    const float* xr = x + (size_t)row * 1024;
    float4 xv = ((const float4*)xr)[t];
    float s  = xv.x + xv.y + xv.z + xv.w;
    float s2 = xv.x * xv.x + xv.y * xv.y + xv.z * xv.z + xv.w * xv.w;
    #pragma unroll
    for (int o = 16; o > 0; o >>= 1) {
        s  += __shfl_xor(s,  o, 32);
        s2 += __shfl_xor(s2, o, 32);
    }
    if ((t & 31) == 0) { red[t >> 5] = s; red[8 + (t >> 5)] = s2; }
    __syncthreads();
    float ts = 0.f, ts2 = 0.f;
    #pragma unroll
    for (int i = 0; i < 8; i++) { ts += red[i]; ts2 += red[8 + i]; }
    float mu  = ts * (1.f / 1024.f);
    float var = ts2 * (1.f / 1024.f) - mu * mu;
    float rs  = rsqrtf(var + 1e-5f);

    const float* gr = gb + (size_t)row * 2048;
    float4 g  = ((const float4*)gr)[t];
    float4 be = ((const float4*)(gr + 1024))[t];
    unsigned short o0 = f2bf((xv.x - mu) * rs * (1.f + g.x) + be.x);
    unsigned short o1 = f2bf((xv.y - mu) * rs * (1.f + g.y) + be.y);
    unsigned short o2 = f2bf((xv.z - mu) * rs * (1.f + g.z) + be.z);
    unsigned short o3 = f2bf((xv.w - mu) * rs * (1.f + g.w) + be.w);
    uint2 pk;
    pk.x = (unsigned)o0 | ((unsigned)o1 << 16);
    pk.y = (unsigned)o2 | ((unsigned)o3 << 16);
    ((uint2*)(h + (size_t)row * 1024))[t] = pk;
}

// silu(x) -> bf16, 2 elements per thread (packed dword store)
__global__ void silu_cast_kernel(const float* __restrict__ in,
                                 unsigned short* __restrict__ out, int n)
{
    int i = blockIdx.x * blockDim.x + threadIdx.x;
    int j = i * 2;
    if (j < n) {
        float a = in[j], b = in[j + 1];
        a = a / (1.f + __expf(-a));
        b = b / (1.f + __expf(-b));
        ((unsigned*)out)[i] = (unsigned)f2bf(a) | ((unsigned)f2bf(b) << 16);
    }
}

// ---------------------------------------------------------------------------
extern "C" void kernel_launch(void* const* d_in, const int* in_sizes, int n_in,
                              void* d_out, int out_size, void* d_ws, size_t ws_size,
                              hipStream_t stream)
{
    const float* x          = (const float*)d_in[0];
    const float* cond       = (const float*)d_in[1];
    const float* p1_w       = (const float*)d_in[2];
    const float* p1_b       = (const float*)d_in[3];
    const float* qkv_w      = (const float*)d_in[4];
    const float* attn_out_w = (const float*)d_in[5];
    const float* p2_w       = (const float*)d_in[6];
    const float* p2_b       = (const float*)d_in[7];
    const float* ffn_w1     = (const float*)d_in[8];
    const float* ffn_b1     = (const float*)d_in[9];
    const float* ffn_w2     = (const float*)d_in[10];
    const float* ffn_b2     = (const float*)d_in[11];
    float* out = (float*)d_out;

    const int Mtok = 4096;   // B*S

    char* ws = (char*)d_ws;
    size_t off = 0;
    auto alloc = [&](size_t bytes) -> char* {
        char* p = ws + off;
        off += (bytes + 255) & ~(size_t)255;
        return p;
    };
    unsigned short* cbf  = (unsigned short*)alloc((size_t)Mtok * 512 * 2);
    float*          gbuf = (float*)         alloc((size_t)Mtok * 2048 * 4);
    unsigned short* hbuf = (unsigned short*)alloc((size_t)Mtok * 1024 * 2);
    unsigned short* qkvb = (unsigned short*)alloc((size_t)Mtok * 3072 * 2);
    unsigned short* attb = (unsigned short*)alloc((size_t)Mtok * 1024 * 2);
    float*          x1   = (float*)         alloc((size_t)Mtok * 1024 * 4);
    unsigned short* ffb  = (unsigned short*)alloc((size_t)Mtok * 4096 * 2);
    (void)ws_size; (void)in_sizes; (void)n_in; (void)out_size;

    // 1. silu(cond) -> bf16
    silu_cast_kernel<<<(Mtok * 512 / 2 + 255) / 256, 256, 0, stream>>>(
        cond, cbf, Mtok * 512);
    // 2. gb1 = silu(cond) @ p1_w + p1_b        (4096 x 2048, K=512)
    gemm_bf16_wmma<<<dim3(2048 / BN, Mtok / BM), 128, 0, stream>>>(
        cbf, p1_w, p1_b, nullptr, gbuf, Mtok, 2048, 512, 0);
    // 3. h = adaln(x, gb1) -> bf16
    adaln_kernel<<<Mtok, 256, 0, stream>>>(x, gbuf, hbuf);
    // 4. qkv = h @ qkv_w -> bf16               (4096 x 3072, K=1024)
    gemm_bf16_wmma<<<dim3(3072 / BN, Mtok / BM), 128, 0, stream>>>(
        hbuf, qkv_w, nullptr, nullptr, qkvb, Mtok, 3072, 1024, 1);
    // 5. attention -> bf16 (B,S,1024)
    attn_wmma<<<1024, 128, 0, stream>>>(qkvb, attb);
    // 6. x1 = x + attn @ attn_out_w            (4096 x 1024, K=1024)
    gemm_bf16_wmma<<<dim3(1024 / BN, Mtok / BM), 128, 0, stream>>>(
        attb, attn_out_w, nullptr, x, x1, Mtok, 1024, 1024, 0);
    // 7. gb2 = silu(cond) @ p2_w + p2_b
    gemm_bf16_wmma<<<dim3(2048 / BN, Mtok / BM), 128, 0, stream>>>(
        cbf, p2_w, p2_b, nullptr, gbuf, Mtok, 2048, 512, 0);
    // 8. h2 = adaln(x1, gb2) -> bf16
    adaln_kernel<<<Mtok, 256, 0, stream>>>(x1, gbuf, hbuf);
    // 9. ff = gelu(h2 @ ffn_w1 + b1) -> bf16   (4096 x 4096, K=1024)
    gemm_bf16_wmma<<<dim3(4096 / BN, Mtok / BM), 128, 0, stream>>>(
        hbuf, ffn_w1, ffn_b1, nullptr, ffb, Mtok, 4096, 1024, 1 | 2);
    // 10. out = x1 + ff @ ffn_w2 + b2          (4096 x 1024, K=4096)
    gemm_bf16_wmma<<<dim3(1024 / BN, Mtok / BM), 128, 0, stream>>>(
        ffb, ffn_w2, ffn_b2, x1, out, Mtok, 1024, 4096, 0);
}
